// OneLayer_LSTM_37632503447566
// MI455X (gfx1250) — matrix operand order repaired and checked by reading, not compiled, not run on hardware
//
#include <hip/hip_runtime.h>
#include <hip/hip_bf16.h>

typedef __bf16 bf16_t;
typedef __attribute__((ext_vector_type(16))) __bf16 v16bf;
typedef __attribute__((ext_vector_type(8)))  __bf16 v8bf;
typedef __attribute__((ext_vector_type(8)))  float  v8f;
typedef __attribute__((ext_vector_type(4)))  unsigned int u32x4;
typedef __attribute__((ext_vector_type(8)))  unsigned int u32x8;

#define SEQ_L 512
#define BATCH 64
#define CDIM  1024
#define HDIM  1024
#define NGATE 4096          // 4*H
#define KTOT  2048          // H + C (rows 0..1023 = h (W_hh), 1024..2047 = x (W_ih))

// workspace byte offsets
#define WS_WT   0                       // bf16 [4096][2048]  = 16 MB gate-interleaved W^T
#define WS_C    16777216                // f32  [64][1024]    cell state
#define WS_H0   (WS_C + 262144)         // bf16 [64][1024]    hidden (ping)
#define WS_H1   (WS_H0 + 131072)        // bf16 [64][1024]    hidden (pong)

// LDS byte offsets (dynamic, 320 KB total = WGP max)
#define LDS_AH  0                       // bf16 [64][1024] h panel  (reused as f32 gates)
#define LDS_AX  131072                  // bf16 [64][1024] x panel
#define LDS_B   262144                  // bf16 [32][1024] W slice (per K-phase)
#define LDS_TOT 327680

// ---- CDNA5 async global->LDS, GVS form (SGPR base + 32-bit lane offset) ----
__device__ __forceinline__ void async_g2l_b128(unsigned lds_addr, unsigned voff,
                                               const void* sbase) {
    asm volatile("global_load_async_to_lds_b128 %0, %1, %2"
                 :: "v"(lds_addr), "v"(voff), "s"(sbase) : "memory");
}
__device__ __forceinline__ void wait_async0() {
    asm volatile("s_wait_asynccnt 0x0" ::: "memory");
}

// ---- CDNA5 Tensor Data Mover: 2D tile (bf16) global -> LDS -----------------
// D# group0: count=1, lds_addr, global_addr[56:0], type=2.
// D# group1: data_size=2B, tensor=cols x rows, tile=cols x rows, dim0 stride.
__device__ __forceinline__ void tdm_load_2d_bf16(unsigned lds_addr, const void* gaddr,
                                                 unsigned rows, unsigned cols,
                                                 unsigned row_stride_elems) {
    unsigned long long ga = (unsigned long long)(uintptr_t)gaddr;
    u32x4 g0;
    g0[0] = 1u;                                               // count=1, no gather
    g0[1] = lds_addr;                                         // lds_addr[31:0]
    g0[2] = (unsigned)ga;                                     // global_addr[31:0]
    g0[3] = ((unsigned)(ga >> 32) & 0x01FFFFFFu) | (2u << 30); // addr[56:32] | type=2
    u32x8 g1;
    g1[0] = 1u << 16;                                         // data_size = 1 (2 bytes)
    g1[1] = (cols & 0xFFFFu) << 16;                           // tensor_dim0[15:0]
    g1[2] = (cols >> 16) | ((rows & 0xFFFFu) << 16);          // dim0[31:16] | dim1[15:0]
    g1[3] = (rows >> 16) | ((cols & 0xFFFFu) << 16);          // dim1[31:16] | tile_dim0
    g1[4] = rows & 0xFFFFu;                                   // tile_dim1 (tile_dim2=0)
    g1[5] = row_stride_elems;                                 // tensor_dim0_stride[31:0]
    g1[6] = 0u;                                               // stride[47:32] | dim1_stride lo
    g1[7] = 0u;
    asm volatile("tensor_load_to_lds %0, %1" :: "s"(g0), "s"(g1) : "memory");
}
__device__ __forceinline__ void wait_tensor0() {
    __builtin_amdgcn_s_wait_tensorcnt(0);
}

// ---------------------------------------------------------------------------
// WT[n'][k] bf16, n' = 4*j + gate (gate-interleaved), K = [W_hh ; W_ih].
// ---------------------------------------------------------------------------
__global__ void lstm_prep_weights(const float* __restrict__ W_ih,
                                  const float* __restrict__ W_hh,
                                  bf16_t* __restrict__ WT) {
    int gid = blockIdx.x * 256 + threadIdx.x;        // 4096*2048
    int np  = gid >> 11;
    int k   = gid & 2047;
    int j    = np >> 2;
    int gate = np & 3;
    int col  = gate * HDIM + j;
    float v = (k < HDIM) ? W_hh[k * NGATE + col]
                         : W_ih[(k - HDIM) * NGATE + col];
    WT[gid] = (bf16_t)v;
}

__global__ void lstm_zero_state(float* __restrict__ c, bf16_t* __restrict__ h0) {
    int gid = blockIdx.x * 256 + threadIdx.x;        // 65536
    c[gid]  = 0.0f;
    h0[gid] = (bf16_t)0.0f;
}

// ---------------------------------------------------------------------------
// One LSTM timestep: 128 WGs x 256 threads (8 wave32). WG owns n' in
// [wg*32, wg*32+32)  <=>  hidden units j in [wg*8, wg*8+8), all 4 gates.
// ---------------------------------------------------------------------------
__launch_bounds__(256)
__global__ void lstm_step(const float* __restrict__ x,      // [L,B,C]
                          const float* __restrict__ bias,   // [4H]
                          const bf16_t* __restrict__ WT,    // [4096][2048]
                          const bf16_t* __restrict__ hin,
                          bf16_t* __restrict__ hout,
                          float* __restrict__ cst,
                          float* __restrict__ out,
                          int t) {
    extern __shared__ char smem[];
    bf16_t* sAh = (bf16_t*)(smem + LDS_AH);
    bf16_t* sAx = (bf16_t*)(smem + LDS_AX);
    bf16_t* sB  = (bf16_t*)(smem + LDS_B);
    const unsigned ldsbase = (unsigned)(uintptr_t)smem;   // flat->LDS: addr[31:0]

    const int tid  = threadIdx.x;
    const int wg   = blockIdx.x;                // 0..127
    const int wave = tid >> 5;
    const int lane = tid & 31;
    const int lrow = lane & 15;
    const int hsel = lane >> 4;
    const int mt   = wave & 3;                  // m-tile (rows mt*16..)
    const int nt   = wave >> 2;                 // n-tile (cols nt*16..)

    const float*  xt    = x + (size_t)t * (BATCH * CDIM);
    const bf16_t* wbase = WT + (size_t)wg * 32 * KTOT;

    // ---- (1a) TDM: W phase-0 slice, 32 rows x 1024 bf16, stride 2048 ----
    if (tid == 0) {
        tdm_load_2d_bf16(ldsbase + LDS_B, wbase, 32u, 1024u, (unsigned)KTOT);
    }
    // ---- (1b) async DMA: h panel (128 KB contiguous), GVS addressing ----
    #pragma unroll
    for (int it = 0; it < 32; ++it) {
        unsigned idx = (unsigned)(it * 256 + tid);           // 8192 x 16B
        async_g2l_b128(ldsbase + LDS_AH + idx * 16u, idx * 16u, hin);
    }

    // ---- (2) overlap: convert x_t f32->bf16 into sAx while DMA flies ----
    #pragma unroll
    for (int it = 0; it < 32; ++it) {
        int idx = it * 256 + tid;                            // 8 floats / chunk
        const float4* s4 = (const float4*)xt + (size_t)idx * 2;
        float4 a0 = s4[0];
        float4 a1 = s4[1];
        v8bf o;
        o[0] = (bf16_t)a0.x; o[1] = (bf16_t)a0.y;
        o[2] = (bf16_t)a0.z; o[3] = (bf16_t)a0.w;
        o[4] = (bf16_t)a1.x; o[5] = (bf16_t)a1.y;
        o[6] = (bf16_t)a1.z; o[7] = (bf16_t)a1.w;
        ((v8bf*)sAx)[idx] = o;
    }

    // ---- (3) prefetch next timestep's x panel into L2 (8 WGs cover 256 KB) ----
    if (wg < 8 && t + 1 < SEQ_L) {
        const char* p = (const char*)(x + (size_t)(t + 1) * (BATCH * CDIM))
                      + (size_t)(wg * 256 + tid) * 128;
        __builtin_prefetch(p, 0, 0);
    }

    wait_async0();
    if (tid == 0) wait_tensor0();
    __syncthreads();

    v8f acc = {};

    // ---- (4) phase 0: gates += h @ W_hh  (K = 0..1023) ----
    {
        const bf16_t* aBase = sAh + (mt * 16 + lrow) * 1024 + hsel * 16;
        const bf16_t* bBase = sB  + (nt * 16 + lrow) * 1024 + hsel * 16;
        #pragma unroll 4
        for (int kk = 0; kk < 1024; kk += 32) {
            v8bf alo = *(const v8bf*)(aBase + kk);
            v8bf ahi = *(const v8bf*)(aBase + kk + 8);
            v16bf a = __builtin_shufflevector(alo, ahi,
                        0, 1, 2, 3, 4, 5, 6, 7, 8, 9, 10, 11, 12, 13, 14, 15);
            v16bf b = *(const v16bf*)(bBase + kk);
            acc = __builtin_amdgcn_wmma_f32_16x16x32_bf16(
                      false, a, false, b, (short)0, acc, false, false);
        }
    }
    __syncthreads();                     // all waves done reading sB

    // ---- (5) TDM: restage W phase-1 slice (K = 1024..2047) ----
    if (tid == 0) {
        tdm_load_2d_bf16(ldsbase + LDS_B, wbase + 1024, 32u, 1024u, (unsigned)KTOT);
        wait_tensor0();
    }
    __syncthreads();

    // ---- (6) phase 1: gates += x_t @ W_ih  (K = 1024..2047) ----
    {
        const bf16_t* aBase = sAx + (mt * 16 + lrow) * 1024 + hsel * 16;
        const bf16_t* bBase = sB  + (nt * 16 + lrow) * 1024 + hsel * 16;
        #pragma unroll 4
        for (int kk = 0; kk < 1024; kk += 32) {
            v8bf alo = *(const v8bf*)(aBase + kk);
            v8bf ahi = *(const v8bf*)(aBase + kk + 8);
            v16bf a = __builtin_shufflevector(alo, ahi,
                        0, 1, 2, 3, 4, 5, 6, 7, 8, 9, 10, 11, 12, 13, 14, 15);
            v16bf b = *(const v16bf*)(bBase + kk);
            acc = __builtin_amdgcn_wmma_f32_16x16x32_bf16(
                      false, a, false, b, (short)0, acc, false, false);
        }
    }
    __syncthreads();                     // all waves done with sAh/sAx/sB

    // ---- (7) scatter accumulators to LDS (reuse sAh region as f32 gates) ----
    float* gts = (float*)smem;           // [64][32]
    {
        int nloc  = nt * 16 + lrow;      // C/D layout: lane -> N
        int mbase = mt * 16 + hsel * 8;  // VGPR r -> M (+8 upper half)
        #pragma unroll
        for (int r = 0; r < 8; ++r)
            gts[(mbase + r) * 32 + nloc] = acc[r];
    }
    __syncthreads();

    // ---- (8) fused LSTM cell update for this WG's (m, j) slice ----
    for (int p = tid; p < 512; p += 256) {
        int m  = p >> 3;
        int jl = p & 7;
        int j  = wg * 8 + jl;
        int gb = m * 32 + jl * 4;
        float gi = gts[gb + 0] + bias[j];
        float gf = gts[gb + 1] + bias[HDIM + j];
        float gg = gts[gb + 2] + bias[2 * HDIM + j];
        float go = gts[gb + 3] + bias[3 * HDIM + j];
        float ig = 1.0f / (1.0f + __expf(-gi));
        float fg = 1.0f / (1.0f + __expf(-gf));
        float gv = tanhf(gg);
        float og = 1.0f / (1.0f + __expf(-go));
        int   sidx = m * HDIM + j;
        float cn = fg * cst[sidx] + ig * gv;
        float hn = og * tanhf(cn);
        cst[sidx] = cn;
        out[(size_t)t * (BATCH * HDIM) + sidx] = hn;
        hout[sidx] = (bf16_t)hn;
        if (t == SEQ_L - 1) {
            out[(size_t)SEQ_L * BATCH * HDIM + sidx] = hn;                  // h_f
            out[(size_t)SEQ_L * BATCH * HDIM + BATCH * HDIM + sidx] = cn;   // c_f
        }
    }
}

extern "C" void kernel_launch(void* const* d_in, const int* in_sizes, int n_in,
                              void* d_out, int out_size, void* d_ws, size_t ws_size,
                              hipStream_t stream) {
    const float* x    = (const float*)d_in[0];
    const float* W_ih = (const float*)d_in[1];
    const float* W_hh = (const float*)d_in[2];
    const float* bias = (const float*)d_in[3];
    float* out = (float*)d_out;

    char* ws = (char*)d_ws;
    bf16_t* WT  = (bf16_t*)(ws + WS_WT);
    float*  cst = (float*)(ws + WS_C);
    bf16_t* h0  = (bf16_t*)(ws + WS_H0);
    bf16_t* h1  = (bf16_t*)(ws + WS_H1);

    lstm_prep_weights<<<(NGATE * KTOT) / 256, 256, 0, stream>>>(W_ih, W_hh, WT);
    lstm_zero_state<<<(BATCH * HDIM) / 256, 256, 0, stream>>>(cst, h0);

    for (int t = 0; t < SEQ_L; ++t) {
        bf16_t* hin  = (t & 1) ? h1 : h0;
        bf16_t* hout = (t & 1) ? h0 : h1;
        lstm_step<<<128, 256, LDS_TOT, stream>>>(x, bias, WT, hin, hout, cst, out, t);
    }
}
